// GraphAttentionLayer_26216480375068
// MI455X (gfx1250) — matrix operand order repaired and checked by reading, not compiled, not run on hardware
//
#include <hip/hip_runtime.h>
#include <hip/hip_fp16.h>

typedef __attribute__((ext_vector_type(16))) _Float16 v16h;
typedef __attribute__((ext_vector_type(8)))  _Float16 v8h;
typedef __attribute__((ext_vector_type(8)))  float    v8f;
typedef __attribute__((ext_vector_type(4)))  float    v4f;
typedef __attribute__((ext_vector_type(8)))  unsigned v8u;

#define NN      8192
#define FDIM    128
#define GAT_ALPHA 0.2f

// ---------------- WMMA helpers (CDNA5 ISA 7.12.2 layouts) ----------------

__device__ __forceinline__ v8f wmma16x16x32(v16h a, v16h b, v8f c) {
  return __builtin_amdgcn_wmma_f32_16x16x32_f16(
      /*neg_a*/false, a, /*neg_b*/false, b, /*c_mod*/(short)0, c,
      /*reuse_a*/false, /*reuse_b*/false);
}

// A fragment 16x32 f16 from fp32 row-major. Lane l: row m=l&15, k-group g=l>>4.
//   halves [0..7]  = K = kb + 8*g + 0..7
//   halves [8..15] = K = kb + 16 + 8*g + 0..7
__device__ __forceinline__ v16h load_A_f32(const float* __restrict__ row, int kb, int lane) {
  const int g = (lane >> 4) << 3;
  const float* p0 = row + kb + g;
  const float* p1 = p0 + 16;
  v4f x0 = *(const v4f*)(p0);
  v4f x1 = *(const v4f*)(p0 + 4);
  v4f x2 = *(const v4f*)(p1);
  v4f x3 = *(const v4f*)(p1 + 4);
  v16h a;
#pragma unroll
  for (int i = 0; i < 4; ++i) {
    a[i]      = (_Float16)x0[i];
    a[4 + i]  = (_Float16)x1[i];
    a[8 + i]  = (_Float16)x2[i];
    a[12 + i] = (_Float16)x3[i];
  }
  return a;
}

// Signed + abs fragments from one read of a {-1,0,1} fp32 row.
// abs is 8x v_and_b32 on the packed halves (no compares, no selects).
__device__ __forceinline__ void load_A_signed_abs(const float* __restrict__ row, int kb,
                                                  int lane, v16h& as, v16h& aa) {
  as = load_A_f32(row, kb, lane);
  v8u u = __builtin_bit_cast(v8u, as);
  u = u & 0x7fff7fffu;
  aa = __builtin_bit_cast(v16h, u);
}

// B fragment 32x16 f16 from transposed f16 buffer bT[n*ld + k].
// Lane l: col n=n0+(l&15), halves [0..7] K = kb+8*(l>>4)+0..7, [8..15] K = kb+16+8*(l>>4)+0..7.
__device__ __forceinline__ v16h load_B(const _Float16* __restrict__ bT, int ld,
                                       int n0, int kb, int lane) {
  const int n  = n0 + (lane & 15);
  const int ko = kb + ((lane >> 4) << 3);
  const _Float16* p = bT + (size_t)n * ld + ko;
  v8h lo = *(const v8h*)(p);
  v8h hi = *(const v8h*)(p + 16);
  return __builtin_shufflevector(lo, hi, 0,1,2,3,4,5,6,7,8,9,10,11,12,13,14,15);
}

// C/D 16x16 f32: lane l col n=l&15, VGPR r -> row (l>>4)*8 + r.
__device__ __forceinline__ void store_C_f32(float* __restrict__ out, int ld,
                                            int i0, int n0, int lane, v8f c) {
  const int n  = n0 + (lane & 15);
  const int mb = i0 + ((lane >> 4) << 3);
#pragma unroll
  for (int r = 0; r < 8; ++r)
    out[(size_t)(mb + r) * ld + n] = c[r];
}

// ---------------- Kernel 0: W (128x128 f32) -> WT f16 (n-major) ---------------
__global__ void k_wt(const float* __restrict__ W, _Float16* __restrict__ WT) {
  for (int i = threadIdx.x; i < FDIM * FDIM; i += blockDim.x) {
    const int n = i >> 7, k = i & 127;
    WT[(size_t)n * FDIM + k] = (_Float16)W[(size_t)k * FDIM + n];
  }
}

// ---------------- Kernel 1: hw = h @ W, stored transposed f16 hwT[f][node] ----
// grid.x = NN/16, block = 128 (4 waves x 32 features)
__global__ void k_hw(const float* __restrict__ h, const _Float16* __restrict__ WT,
                     _Float16* __restrict__ hwT) {
  const int lane = threadIdx.x & 31;
  const int wave = threadIdx.x >> 5;
  const int i0 = blockIdx.x << 4;
  const int n0 = wave << 5;
  const float* rowp = h + (size_t)(i0 + (lane & 15)) * FDIM;
  v8f c0 = {}, c1 = {};
#pragma unroll
  for (int kb = 0; kb < FDIM; kb += 32) {
    v16h a  = load_A_f32(rowp, kb, lane);
    v16h b0 = load_B(WT, FDIM, n0, kb, lane);
    v16h b1 = load_B(WT, FDIM, n0 + 16, kb, lane);
    c0 = wmma16x16x32(a, b0, c0);
    c1 = wmma16x16x32(a, b1, c1);
  }
  const int mb = i0 + ((lane >> 4) << 3);
  v8h t0, t1;
#pragma unroll
  for (int r = 0; r < 8; ++r) { t0[r] = (_Float16)c0[r]; t1[r] = (_Float16)c1[r]; }
  *(v8h*)(hwT + (size_t)(n0 + (lane & 15)) * NN + mb)      = t0;
  *(v8h*)(hwT + (size_t)(n0 + 16 + (lane & 15)) * NN + mb) = t1;
}

// ---------------- Kernel 2: h_plus/h_minus via signed/abs identity ------------
// h_plus = (|A|@hw + A@hw)/2, h_minus = (|A|@hw - A@hw)/2.
// grid.x = NN/16, block = 128: 4 waves per 16-row block, 32 features per wave
// (2048 waves machine-wide for HBM latency hiding; A-side VALU is not the
// bottleneck, HBM streaming is).
__global__ void k_agg(const float* __restrict__ node_adj, const _Float16* __restrict__ hwT,
                      float* __restrict__ h_plus, float* __restrict__ h_minus) {
  const int lane = threadIdx.x & 31;
  const int wave = threadIdx.x >> 5;
  const int i0 = blockIdx.x << 4;
  const int n0 = wave << 5;
  const float* rowp = node_adj + (size_t)(i0 + (lane & 15)) * NN;
  v8f cs0 = {}, cs1 = {};   // signed accumulators
  v8f ca0 = {}, ca1 = {};   // abs accumulators
  for (int kb = 0; kb < NN; kb += 32) {
    __builtin_prefetch(rowp + kb + 1024, 0, 1);   // global_prefetch_b8 stream-ahead
    v16h as, aa;
    load_A_signed_abs(rowp, kb, lane, as, aa);
    v16h b0 = load_B(hwT, NN, n0, kb, lane);
    v16h b1 = load_B(hwT, NN, n0 + 16, kb, lane);
    cs0 = wmma16x16x32(as, b0, cs0);
    ca0 = wmma16x16x32(aa, b0, ca0);
    cs1 = wmma16x16x32(as, b1, cs1);
    ca1 = wmma16x16x32(aa, b1, ca1);
  }
  store_C_f32(h_plus,  FDIM, i0, n0,      lane, (ca0 + cs0) * 0.5f);
  store_C_f32(h_minus, FDIM, i0, n0,      lane, (ca0 - cs0) * 0.5f);
  store_C_f32(h_plus,  FDIM, i0, n0 + 16, lane, (ca1 + cs1) * 0.5f);
  store_C_f32(h_minus, FDIM, i0, n0 + 16, lane, (ca1 - cs1) * 0.5f);
}

// ---------------- Kernel 3: EP/EM = exp(leaky_relu(dot products)) -------------
// Raw exp is safe here (scores are O(10)); the softmax max-shift cancels exactly.
__global__ void k_e(const float* __restrict__ h_plus, const float* __restrict__ h_minus,
                    const float* __restrict__ a,
                    float* __restrict__ EP, float* __restrict__ EM) {
  const int i = blockIdx.x * blockDim.x + threadIdx.x;
  if (i >= NN) return;
  const float* hp = h_plus + (size_t)i * FDIM;
  const float* hm = h_minus + (size_t)i * FDIM;
  float pa1 = 0.f, ma2 = 0.f, ma1 = 0.f, pa2 = 0.f;
#pragma unroll 4
  for (int k = 0; k < FDIM; k += 4) {
    v4f p  = *(const v4f*)(hp + k);
    v4f m  = *(const v4f*)(hm + k);
    v4f a1 = *(const v4f*)(a + k);
    v4f a2 = *(const v4f*)(a + FDIM + k);
#pragma unroll
    for (int j = 0; j < 4; ++j) {
      pa1 += p[j] * a1[j];  ma2 += m[j] * a2[j];
      ma1 += m[j] * a1[j];  pa2 += p[j] * a2[j];
    }
  }
  float ep = pa1 + ma2; ep = ep > 0.f ? ep : GAT_ALPHA * ep;
  float em = ma1 + pa2; em = em > 0.f ? em : GAT_ALPHA * em;
  EP[i] = __expf(ep);
  EM[i] = __expf(em);
}

// ---------------- Kernel 4: fused denominator + attention GEMM ----------------
// Pre-pass: s[row] = sum_j (edge>0 ? EP[j] : edge<0 ? EM[j] : 0)  (raw softmax
// denominator; the row block stays L2-hot for the main pass). Main pass:
// coeff = sign * (EP|EM)[j] / s built into f16 A fragments, WMMA'd against hwT.
// grid.x = NN/16, block = 128: 4 waves per 16-row block, 32 features per wave.
__global__ void k_final(const float* __restrict__ edge_adj,
                        const float* __restrict__ EP, const float* __restrict__ EM,
                        const _Float16* __restrict__ hwT, float* __restrict__ out) {
  const int lane = threadIdx.x & 31;
  const int wave = threadIdx.x >> 5;
  const int i0 = blockIdx.x << 4;
  const int n0 = wave << 5;
  const int g8 = (lane >> 4) << 3;
  const float* er = edge_adj + (size_t)(i0 + (lane & 15)) * NN;

  // ---- denominator pre-pass (lane pair l / l+16 covers the whole row) ----
  float s = 0.f;
  for (int jb = 0; jb < NN; jb += 16) {
    const int j = jb + g8;
    __builtin_prefetch(er + j + 2048, 0, 1);
    v4f e0 = *(const v4f*)(er + j);
    v4f e1 = *(const v4f*)(er + j + 4);
    v4f p0 = *(const v4f*)(EP + j);
    v4f p1 = *(const v4f*)(EP + j + 4);
    v4f m0 = *(const v4f*)(EM + j);
    v4f m1 = *(const v4f*)(EM + j + 4);
#pragma unroll
    for (int i = 0; i < 4; ++i) {
      s += (e0[i] > 0.f) ? p0[i] : ((e0[i] < 0.f) ? m0[i] : 0.f);
      s += (e1[i] > 0.f) ? p1[i] : ((e1[i] < 0.f) ? m1[i] : 0.f);
    }
  }
  s += __shfl_xor(s, 16, 32);
  const float rinv = (s > 0.f) ? (1.0f / s) : 0.f;  // empty row -> all-zero output

  // ---- main pass: coefficient fragments + WMMA ----
  v8f c0 = {}, c1 = {};
  for (int kb = 0; kb < NN; kb += 32) {
    v16h acf;
#pragma unroll
    for (int hf = 0; hf < 2; ++hf) {
      const int kk = kb + g8 + (hf << 4);
      v4f e0 = *(const v4f*)(er + kk);
      v4f e1 = *(const v4f*)(er + kk + 4);
      v4f p0 = *(const v4f*)(EP + kk);
      v4f p1 = *(const v4f*)(EP + kk + 4);
      v4f m0 = *(const v4f*)(EM + kk);
      v4f m1 = *(const v4f*)(EM + kk + 4);
#pragma unroll
      for (int i = 0; i < 4; ++i) {
        float v0 = ((e0[i] > 0.f) ? p0[i] : ((e0[i] < 0.f) ? -m0[i] : 0.f)) * rinv;
        float v1 = ((e1[i] > 0.f) ? p1[i] : ((e1[i] < 0.f) ? -m1[i] : 0.f)) * rinv;
        acf[(hf << 3) + i]     = (_Float16)v0;
        acf[(hf << 3) + 4 + i] = (_Float16)v1;
      }
    }
    v16h b0 = load_B(hwT, NN, n0, kb, lane);
    v16h b1 = load_B(hwT, NN, n0 + 16, kb, lane);
    c0 = wmma16x16x32(acf, b0, c0);
    c1 = wmma16x16x32(acf, b1, c1);
  }
  store_C_f32(out, FDIM, i0, n0,      lane, c0);
  store_C_f32(out, FDIM, i0, n0 + 16, lane, c1);
}

// ---------------- Host-side launch ----------------
extern "C" void kernel_launch(void* const* d_in, const int* in_sizes, int n_in,
                              void* d_out, int out_size, void* d_ws, size_t ws_size,
                              hipStream_t stream) {
  const float* h        = (const float*)d_in[0];
  const float* node_adj = (const float*)d_in[1];
  const float* edge_adj = (const float*)d_in[2];
  const float* W        = (const float*)d_in[3];
  const float* a        = (const float*)d_in[4];
  float* out = (float*)d_out;

  char* ws = (char*)d_ws;
  _Float16* WT   = (_Float16*)ws;  ws += (size_t)FDIM * FDIM * sizeof(_Float16);   // 32 KB
  _Float16* hwT  = (_Float16*)ws;  ws += (size_t)FDIM * NN   * sizeof(_Float16);   // 2 MB
  float* h_plus  = (float*)ws;     ws += (size_t)NN * FDIM * sizeof(float);        // 4 MB
  float* h_minus = (float*)ws;     ws += (size_t)NN * FDIM * sizeof(float);        // 4 MB
  float* EP      = (float*)ws;     ws += (size_t)NN * sizeof(float);
  float* EM      = (float*)ws;     ws += (size_t)NN * sizeof(float);

  k_wt   <<<1, 256, 0, stream>>>(W, WT);
  k_hw   <<<NN / 16, 128, 0, stream>>>(h, WT, hwT);
  k_agg  <<<NN / 16, 128, 0, stream>>>(node_adj, hwT, h_plus, h_minus);
  k_e    <<<NN / 256, 256, 0, stream>>>(h_plus, h_minus, a, EP, EM);
  k_final<<<NN / 16, 128, 0, stream>>>(edge_adj, EP, EM, hwT, out);
}
